// ScaledDotProductAttention_5188320493986
// MI455X (gfx1250) — matrix operand order repaired and checked by reading, not compiled, not run on hardware
//
#include <hip/hip_runtime.h>

typedef __attribute__((ext_vector_type(16))) _Float16 v16h;
typedef __attribute__((ext_vector_type(8)))  float    v8f;

#define WMMA_F16(a, b, c) \
  __builtin_amdgcn_wmma_f32_16x16x32_f16(false, (a), false, (b), (short)0, (c), false, false)

// ---- WMMA operand loaders -------------------------------------------------
// A-matrix 16x32 f16 layout (ISA 7.12.2): lane l holds row M=l&15;
// elements e=0..7 -> K = kbase + 8*half + e ; e=8..15 -> K = kbase + 16 + 8*half + (e-8)
static __device__ __forceinline__ v16h load_A_f32(const float* p, int ld, int row,
                                                  int kbase, int half) {
  const float* p0 = p + (size_t)row * ld + kbase + 8 * half;
  const float* p1 = p0 + 16;
  v16h a;
#pragma unroll
  for (int e = 0; e < 8; ++e) a[e] = (_Float16)p0[e];
#pragma unroll
  for (int e = 0; e < 8; ++e) a[8 + e] = (_Float16)p1[e];
  return a;
}

static __device__ __forceinline__ v16h load_A_f16(const _Float16* p, int ld, int row,
                                                  int kbase, int half) {
  const _Float16* p0 = p + (size_t)row * ld + kbase + 8 * half;
  const _Float16* p1 = p0 + 16;
  v16h a;
#pragma unroll
  for (int e = 0; e < 8; ++e) a[e] = p0[e];
#pragma unroll
  for (int e = 0; e < 8; ++e) a[8 + e] = p1[e];
  return a;
}

// B-matrix 32x16 f16, where B[k][n] = M[n0+n][kbase+k] (i.e. B = M^T chunk).
// Lane layout: element e<8 holds K row 8*half + e, e>=8 holds K row 16 + 8*half + (e-8), column n = lane&15.
// Per lane this is two 8-float reads of M's row (n0+n).
static __device__ __forceinline__ v16h load_Bt_f32(const float* p, int ld, int n0,
                                                   int kbase, int lane) {
  const int n = lane & 15, half = lane >> 4;
  const float* q = p + (size_t)(n0 + n) * ld + kbase + 8 * half;
  v16h b;
#pragma unroll
  for (int e = 0; e < 16; ++e) b[e] = (_Float16)q[e < 8 ? e : e + 8];
  return b;
}

// ---- Kernel 1: causal flash attention, one wave per 32-row query tile -----
// M=32 per wave: each K/V B-operand is reused by two A tiles (2x WMMA per load).
__global__ __launch_bounds__(256) void fa_causal_wmma_kernel(
    const float* __restrict__ qg, const float* __restrict__ kg,
    const float* __restrict__ vg, _Float16* __restrict__ attn /*[B*S,1024] f16*/) {
  __shared__ _Float16 Vts[8][64][32];  // per-wave V block transposed: Vt[d][key]
  __shared__ _Float16 Pls[8][32][32];  // per-wave P tile row-major [row][key]

  const int lane = threadIdx.x & 31;
  const int wid  = threadIdx.x >> 5;
  const int gw   = blockIdx.x * 8 + wid;  // 0..2047
  const int bh   = gw >> 6;               // B*H = 32, S/32 = 64 tiles each
  const int qt   = gw & 63;
  const int q0   = qt * 32;
  const int bidx = bh >> 4;
  const int h    = bh & 15;
  const int n    = lane & 15;
  const int half = lane >> 4;

  const size_t base = (size_t)bh * 2048 * 64;
  const float* Q = qg + base;
  const float* K = kg + base;
  const float* V = vg + base;

  // Q tiles in A format: [row tile][d chunk]
  v16h aq[2][2];
#pragma unroll
  for (int rt = 0; rt < 2; ++rt)
#pragma unroll
    for (int c = 0; c < 2; ++c)
      aq[rt][c] = load_A_f32(Q, 64, q0 + 16 * rt + (lane & 15), 32 * c, half);

  v8f o[2][4] = {};
  float mrow[2][8], lrow[2][8];
#pragma unroll
  for (int rt = 0; rt < 2; ++rt)
#pragma unroll
    for (int j = 0; j < 8; ++j) { mrow[rt][j] = -1e30f; lrow[rt][j] = 0.0f; }

  const float scale = 0.03125f;  // 1/sqrt(d_model=1024), matching the reference
  const int nkb = (q0 >> 5) + 1; // keys <= q0+31 -> (q0/32)+1 blocks of 32

  _Float16* Vt = &Vts[wid][0][0];
  _Float16* Pl = &Pls[wid][0][0];

  for (int kb = 0; kb < nkb; ++kb) {
    const int kk0 = kb * 32;

    // Stage 32x64 V block transposed into LDS: one key row per lane.
    {
      const float* vr = V + (size_t)(kk0 + lane) * 64;
#pragma unroll
      for (int d = 0; d < 64; ++d) Vt[d * 32 + lane] = (_Float16)vr[d];
    }
    if (kb + 1 < nkb) __builtin_prefetch(K + (size_t)(kk0 + 32) * 64, 0, 1);

    // S = Q K^T : 4 C tiles (2 row tiles x 2 key sub-tiles), B reused across rows.
    v8f s[2][2] = {};
#pragma unroll
    for (int ns = 0; ns < 2; ++ns) {
      const v16h bk0 = load_Bt_f32(K, 64, kk0 + 16 * ns, 0, lane);
      const v16h bk1 = load_Bt_f32(K, 64, kk0 + 16 * ns, 32, lane);
#pragma unroll
      for (int rt = 0; rt < 2; ++rt) {
        s[rt][ns] = WMMA_F16(aq[rt][0], bk0, s[rt][ns]);
        s[rt][ns] = WMMA_F16(aq[rt][1], bk1, s[rt][ns]);
      }
    }

    // Scale + causal mask + online softmax; rescale O and emit P to LDS inline.
#pragma unroll
    for (int rt = 0; rt < 2; ++rt) {
#pragma unroll
      for (int j = 0; j < 8; ++j) {
        const int row = q0 + 16 * rt + j + 8 * half;
        float x0 = s[rt][0][j] * scale;
        float x1 = s[rt][1][j] * scale;
        if (kk0 + n      > row) x0 = -1e30f;
        if (kk0 + 16 + n > row) x1 = -1e30f;
        float rm = fmaxf(x0, x1);
#pragma unroll
        for (int off = 1; off < 16; off <<= 1) rm = fmaxf(rm, __shfl_xor(rm, off, 32));
        const float mnew = fmaxf(mrow[rt][j], rm);
        const float a = __expf(mrow[rt][j] - mnew);
        x0 = __expf(x0 - mnew);
        x1 = __expf(x1 - mnew);
        float rs = x0 + x1;
#pragma unroll
        for (int off = 1; off < 16; off <<= 1) rs += __shfl_xor(rs, off, 32);
        lrow[rt][j] = lrow[rt][j] * a + rs;
        mrow[rt][j] = mnew;
        o[rt][0][j] *= a; o[rt][1][j] *= a;
        o[rt][2][j] *= a; o[rt][3][j] *= a;
        const int r = 16 * rt + j + 8 * half;
        Pl[r * 32 + n]      = (_Float16)x0;
        Pl[r * 32 + 16 + n] = (_Float16)x1;
      }
    }
    asm volatile("s_wait_dscnt 0" ::: "memory");  // DS in-order within wave

    // Reload P in A format for both row tiles.
    v16h pa[2];
#pragma unroll
    for (int rt = 0; rt < 2; ++rt) {
      const _Float16* pr = Pl + (16 * rt + (lane & 15)) * 32;
#pragma unroll
      for (int e = 0; e < 8; ++e) pa[rt][e] = pr[8 * half + e];
#pragma unroll
      for (int e = 0; e < 8; ++e) pa[rt][8 + e] = pr[16 + 8 * half + e];
    }

    // O += P V : B[k][n] = V[kk0+k][dsub+n] = Vt[dsub+n][k]; B reused across rows.
#pragma unroll
    for (int t = 0; t < 4; ++t) {
      v16h bv;
      const _Float16* c = Vt + (16 * t + n) * 32 + 8 * half;
#pragma unroll
      for (int e = 0; e < 16; ++e) bv[e] = c[e < 8 ? e : e + 8];
      o[0][t] = WMMA_F16(pa[0], bv, o[0][t]);
      o[1][t] = WMMA_F16(pa[1], bv, o[1][t]);
    }
  }

  // Normalize and store into permuted layout: d_model index = d*16 + h.
  const size_t trow = (size_t)bidx * 2048 + q0;
#pragma unroll
  for (int rt = 0; rt < 2; ++rt) {
#pragma unroll
    for (int j = 0; j < 8; ++j) {
      const float inv = 1.0f / lrow[rt][j];
      const size_t t = trow + 16 * rt + j + 8 * half;
#pragma unroll
      for (int dt = 0; dt < 4; ++dt)
        attn[t * 1024 + (16 * dt + n) * 16 + h] = (_Float16)(o[rt][dt][j] * inv);
    }
  }
}

// ---- Kernel 2: Y[4096,1024] = X f16 @ W^T + b (f32 out) -------------------
// M=32 per wave: each W B-operand is reused by two A tiles.
__global__ __launch_bounds__(256) void proj_wmma_kernel(
    const _Float16* __restrict__ X, const float* __restrict__ W,
    const float* __restrict__ bias, float* __restrict__ Y) {
  const int lane = threadIdx.x & 31;
  const int wid  = threadIdx.x >> 5;
  const int gw   = blockIdx.x * 8 + wid;  // 0..2047
  const int m0   = (gw >> 4) * 32;        // 128 row tiles of 32
  const int n0   = (gw & 15) * 64;        // 16 col groups of 64
  const int n    = lane & 15;
  const int half = lane >> 4;

  v8f acc[2][4] = {};
  for (int kb = 0; kb < 1024; kb += 32) {
    const v16h a0 = load_A_f16(X, 1024, m0 + (lane & 15), kb, half);
    const v16h a1 = load_A_f16(X, 1024, m0 + 16 + (lane & 15), kb, half);
#pragma unroll
    for (int t = 0; t < 4; ++t) {
      const v16h bw = load_Bt_f32(W, 1024, n0 + 16 * t, kb, lane);
      acc[0][t] = WMMA_F16(a0, bw, acc[0][t]);
      acc[1][t] = WMMA_F16(a1, bw, acc[1][t]);
    }
  }
#pragma unroll
  for (int t = 0; t < 4; ++t) {
    const int oc = n0 + 16 * t + n;
    const float bb = bias[oc];
#pragma unroll
    for (int rt = 0; rt < 2; ++rt)
#pragma unroll
      for (int j = 0; j < 8; ++j)
        Y[(size_t)(m0 + 16 * rt + j + 8 * half) * 1024 + oc] = acc[rt][t][j] + bb;
  }
}

extern "C" void kernel_launch(void* const* d_in, const int* in_sizes, int n_in,
                              void* d_out, int out_size, void* d_ws, size_t ws_size,
                              hipStream_t stream) {
  (void)in_sizes; (void)n_in; (void)out_size; (void)ws_size;
  const float* q = (const float*)d_in[0];
  const float* k = (const float*)d_in[1];
  const float* v = (const float*)d_in[2];
  // d_in[3] is the boolean causal mask; the kernel applies causality analytically.
  const float* W = (const float*)d_in[4];
  const float* b = (const float*)d_in[5];

  _Float16* attn = (_Float16*)d_ws;  // [B*S, 1024] f16 = 8 MB scratch
  float* out = (float*)d_out;        // [B, S, 1024] f32

  fa_causal_wmma_kernel<<<256, 256, 0, stream>>>(q, k, v, attn);
  proj_wmma_kernel<<<256, 256, 0, stream>>>(attn, W, b, out);
}
